// DiffeqSolver_71012989272603
// MI455X (gfx1250) — matrix operand order repaired and checked by reading, not compiled, not run on hardware
//
#include <hip/hip_runtime.h>

typedef __attribute__((ext_vector_type(16))) _Float16 v16h;
typedef __attribute__((ext_vector_type(8)))  _Float16 v8h;
typedef __attribute__((ext_vector_type(8)))  float    v8f;

namespace {
constexpr int S  = 4, Bt = 512, D = 256, H = 1024, T = 64;
constexpr int M  = S * Bt;          // 2048 state rows
constexpr int MT = 16;              // rows per workgroup = one WMMA M-tile
constexpr int NWG = M / MT;         // 128 independent workgroups (no global sync!)
constexpr int NTHREADS = 256;       // 8 x wave32
// padded LDS strides (avoid 64-bank conflicts on 512B/2KB rows)
constexpr int ASTR = 264;           // halfs  -> 528 B/row (16B aligned rows)
constexpr int HSTR = 1032;          // halfs  -> 2064 B/row (16B aligned rows)
constexpr int KSTR = 260;           // floats
constexpr int HLDS_BYTES = MT * HSTR * 2;  // 33024
constexpr int ALDS_BYTES = MT * ASTR * 2;  //  8448
constexpr int KLDS_BYTES = MT * KSTR * 4;  // 16640
constexpr int BLDS_BYTES = H * 4;          //  4096 (b1 table)
// total static LDS = 62208 B
}

// A-matrix 16x32 f16 fragment: lane m = lane&15, half-group hh:
// halfs 0..15 = K[k0 + hh*16 .. +15] (contiguous, matches B loads)
__device__ __forceinline__ v16h make_afrag(const _Float16* rowp, int hh) {
  v8h lo = *(const v8h*)(rowp + hh * 16);
  v8h hi = *(const v8h*)(rowp + hh * 16 + 8);
  return __builtin_shufflevector(lo, hi, 0,1,2,3,4,5,6,7,8,9,10,11,12,13,14,15);
}

__device__ __forceinline__ v8f wmma_f16(v16h a, v16h b, v8f c) {
  // 8 args: (neg_a, A, neg_b, B, c_mod, C, reuse_a, reuse_b)
  return __builtin_amdgcn_wmma_f32_16x16x32_f16(false, a, false, b, (short)0, c,
                                                false, false);
}

// Branch-free tanh: tanh(x) = sign(x) * (1 - 2/(2^(2*log2(e)*|x|) + 1)).
// v_exp_f32 + v_rcp_f32, no EXEC divergence; saturates correctly (rcp(inf)=0).
__device__ __forceinline__ float fast_tanh(float x) {
  float ax = __builtin_fabsf(x);
  float e2 = __builtin_amdgcn_exp2f(ax * 2.88539008177792681f); // 2*log2(e)
  float r  = 1.0f - 2.0f * __builtin_amdgcn_rcpf(e2 + 1.0f);
  return __builtin_copysignf(r, x);
}

// One-shot fp32 -> f16 weight conversion into workspace.
__global__ void cvt_weights_f16(const float* __restrict__ W1,
                                const float* __restrict__ W2,
                                _Float16* __restrict__ W1h,
                                _Float16* __restrict__ W2h, int n) {
  int i = blockIdx.x * blockDim.x + threadIdx.x;
  if (i < n) {
    W1h[i] = (_Float16)W1[i];
    W2h[i] = (_Float16)W2[i];
  }
}

// Each workgroup integrates a private 16-row tile of y through all 63 RK4
// steps. f(y) rows are independent, so no cross-workgroup sync is needed.
// launch_bounds minWavesPerEU=4 caps VGPRs at 256: no extended-VGPR MSB
// addressing, no scratch spills in the WMMA hot loop.
__global__ __launch_bounds__(NTHREADS, 4)
void node_rk4_fused(const float* __restrict__ y0,
                    const float* __restrict__ times,
                    const float* __restrict__ b1,
                    const float* __restrict__ b2,
                    const _Float16* __restrict__ W1h,   // [H][D] row-major f16
                    const _Float16* __restrict__ W2h,   // [D][H] row-major f16
                    float* __restrict__ out)            // [S,B,T,D]
{
  __shared__ __align__(32) char smem[HLDS_BYTES + ALDS_BYTES + KLDS_BYTES + BLDS_BYTES];
  _Float16* hlds = (_Float16*)smem;                          // tanh acts [16][HSTR]
  _Float16* alds = (_Float16*)(smem + HLDS_BYTES);           // A tile f16 [16][ASTR]
  float*    kcur = (float*)(smem + HLDS_BYTES + ALDS_BYTES); // current k [16][KSTR]
  float*    b1ls = (float*)(smem + HLDS_BYTES + ALDS_BYTES + KLDS_BYTES); // b1 [H]

  const int tid  = threadIdx.x;
  const int wv   = tid >> 5;      // wave 0..7
  const int lane = tid & 31;
  const int n16  = lane & 15;     // output column / A row within tile
  const int hh   = lane >> 4;     // lane half-group
  const int r0   = blockIdx.x * MT;

  // Thread `tid` privately owns column d = tid of all 16 rows.
  float yreg[MT];   // y[m][tid]
  float ksreg[MT];  // k1 + 2*k2 + 2*k3 accumulator

  // b1 -> LDS table (read in GEMM1 epilogue with plain ds_load, no movrel).
  #pragma unroll
  for (int i = tid; i < H; i += NTHREADS) b1ls[i] = b1[i];
  // b2: two values per thread, constant indices -> registers.
  const float b2va = b2[wv * 32 + n16];
  const float b2vb = b2[wv * 32 + 16 + n16];

  #pragma unroll
  for (int m = 0; m < MT; ++m) {
    float v = y0[(size_t)(r0 + m) * D + tid];
    yreg[m] = v;
    out[(size_t)(r0 + m) * T * D + tid] = v;   // t = 0 slice
  }
  __syncthreads();   // b1ls visible to all waves

  for (int t = 1; t < T; ++t) {
    const float dt = times[t] - times[t - 1];

    #pragma unroll 1
    for (int fe = 0; fe < 4; ++fe) {
      const float c = (fe == 0) ? 0.0f : (fe == 3 ? dt : 0.5f * dt);

      // ---- phase 1: alds = f16(y + c*k_prev); fold RK4 sum update ----
      #pragma unroll
      for (int m = 0; m < MT; ++m) {
        float kv = 0.0f;
        if (fe != 0) kv = kcur[m * KSTR + tid];
        if (fe == 1)      ksreg[m]  = kv;          // ksum = k1
        else if (fe >= 2) ksreg[m] += 2.0f * kv;   // += 2*k2, += 2*k3
        alds[m * ASTR + tid] = (_Float16)(yreg[m] + c * kv);
      }
      __syncthreads();

      // ---- GEMM1 (16x1024x256) + bias + tanh -> hlds (f16) ----
      // A frags are N-invariant: preload all 8 K-steps once per wave.
      v16h afr[8];
      {
        const _Float16* ap = alds + n16 * ASTR;
        #pragma unroll
        for (int ks = 0; ks < 8; ++ks) afr[ks] = make_afrag(ap + ks * 32, hh);
      }
      // unroll 1: keeps B streaming from L2 (no cross-t hoisting / spills);
      // two independent zero-init accumulation chains per pair (inline C=0).
      #pragma unroll 1
      for (int ntp = 0; ntp < 4; ++ntp) {
        const int n0a = (wv * 8 + 2 * ntp) * 16;
        const int n0b = n0a + 16;
        v8f acca = {};
        v8f accb = {};
        const _Float16* browa = W1h + (size_t)(n0a + n16) * D;
        const _Float16* browb = W1h + (size_t)(n0b + n16) * D;
        // speculative prefetch of next tile-pair's weight rows
        __builtin_prefetch(browa + 32 * D, 0, 1);
        __builtin_prefetch(browb + 32 * D, 0, 1);
        #pragma unroll
        for (int ks = 0; ks < 8; ++ks) {
          v16h bfa = *(const v16h*)(browa + ks * 32 + hh * 16);
          v16h bfb = *(const v16h*)(browb + ks * 32 + hh * 16);
          acca = wmma_f16(afr[ks], bfa, acca);
          accb = wmma_f16(afr[ks], bfb, accb);
        }
        const float bva = b1ls[n0a + n16];
        const float bvb = b1ls[n0b + n16];
        #pragma unroll
        for (int j = 0; j < 8; ++j) {             // C/D: row m = j + 8*hh, col n16
          hlds[(j + 8 * hh) * HSTR + n0a + n16] = (_Float16)fast_tanh(acca[j] + bva);
          hlds[(j + 8 * hh) * HSTR + n0b + n16] = (_Float16)fast_tanh(accb[j] + bvb);
        }
      }
      __syncthreads();

      // ---- GEMM2 (16x256x1024) + bias -> kcur (fp32) ----
      {
        const int n0a = wv * 32, n0b = n0a + 16;  // 2 N-tiles per wave
        v8f acca = {};                            // inline C=0; bias in epilogue
        v8f accb = {};
        const _Float16* ah    = hlds + n16 * HSTR;
        const _Float16* browa = W2h + (size_t)(n0a + n16) * H;
        const _Float16* browb = W2h + (size_t)(n0b + n16) * H;
        #pragma unroll 4
        for (int ks = 0; ks < 32; ++ks) {
          v16h af  = make_afrag(ah + ks * 32, hh);
          v16h bfa = *(const v16h*)(browa + ks * 32 + hh * 16);
          v16h bfb = *(const v16h*)(browb + ks * 32 + hh * 16);
          acca = wmma_f16(af, bfa, acca);
          accb = wmma_f16(af, bfb, accb);
        }
        #pragma unroll
        for (int j = 0; j < 8; ++j) {
          kcur[(j + 8 * hh) * KSTR + n0a + n16] = acca[j] + b2va;
          kcur[(j + 8 * hh) * KSTR + n0b + n16] = accb[j] + b2vb;
        }
      }
      __syncthreads();
    } // fe

    // ---- RK4 combine: y += dt/6 * (k1 + 2k2 + 2k3 + k4); emit slice t ----
    const float s6 = dt * (1.0f / 6.0f);
    #pragma unroll
    for (int m = 0; m < MT; ++m) {
      float yn = yreg[m] + s6 * (ksreg[m] + kcur[m * KSTR + tid]);
      yreg[m] = yn;
      out[((size_t)(r0 + m) * T + t) * D + tid] = yn;
    }
    __syncthreads();
  }
}

extern "C" void kernel_launch(void* const* d_in, const int* in_sizes, int n_in,
                              void* d_out, int out_size, void* d_ws, size_t ws_size,
                              hipStream_t stream) {
  const float* first_point = (const float*)d_in[0];
  const float* times       = (const float*)d_in[1];
  const float* W1          = (const float*)d_in[2];
  const float* b1          = (const float*)d_in[3];
  const float* W2          = (const float*)d_in[4];
  const float* b2          = (const float*)d_in[5];
  float* out = (float*)d_out;

  _Float16* W1h = (_Float16*)d_ws;
  _Float16* W2h = (_Float16*)((char*)d_ws + (size_t)H * D * sizeof(_Float16));

  constexpr int NW = H * D;  // 262144 elements each
  cvt_weights_f16<<<(NW + 255) / 256, 256, 0, stream>>>(W1, W2, W1h, W2h, NW);
  node_rk4_fused<<<NWG, NTHREADS, 0, stream>>>(first_point, times, b1, b2,
                                               W1h, W2h, out);
}
